// MultiHeadLatentAttention_10746008174603
// MI455X (gfx1250) — matrix operand order repaired and checked by reading, not compiled, not run on hardware
//
#include <hip/hip_runtime.h>

// ---------------------------------------------------------------------------
// MLA forward for MI455X (gfx1250): bf16 WMMA GEMMs + flash attention.
// Round 4: pre-transpose weights (and V) once so ALL hot-loop LDS staging is
// contiguous global_load_async_to_lds_b128 — no scalar transpose scatter.
// ---------------------------------------------------------------------------

typedef __bf16 bf16;
typedef __attribute__((ext_vector_type(16))) __bf16 v16bf;
typedef __attribute__((ext_vector_type(8)))  __bf16 v8bf;
typedef __attribute__((ext_vector_type(8)))  float  v8f;
typedef __attribute__((ext_vector_type(4)))  int    v4i_t;

#define AS1 __attribute__((address_space(1)))
#define AS3 __attribute__((address_space(3)))

#define D_MODEL  2048
#define N_HEADS  16
#define D_HEAD   128
#define Q_COMP   1536
#define KV_COMP  512
#define ROPE_DIM 64
#define D_QK     192   // D_HEAD + ROPE_DIM
#define B_SZ     2
#define T_SZ     2048
#define M_ROWS   (B_SZ * T_SZ)   // 4096

#if __has_builtin(__builtin_amdgcn_global_load_async_to_lds_b128) && \
    __has_builtin(__builtin_amdgcn_s_wait_asynccnt)
#define MLA_HAS_ASYNC 1
#else
#define MLA_HAS_ASYNC 0
#endif

// ---------------------------------------------------------------------------
// helpers
// ---------------------------------------------------------------------------
__device__ __forceinline__ v16bf pack16(v8bf lo, v8bf hi) {
    v16bf r;
#pragma unroll
    for (int i = 0; i < 8; ++i) { r[i] = lo[i]; r[i + 8] = hi[i]; }
    return r;
}

// A fragment (16x32 bf16): lane-half holds 16 contiguous K values k0..k0+15
__device__ __forceinline__ v16bf load_frag_interleave8(const bf16* p) {
    v8bf lo = *(const v8bf*)p;
    v8bf hi = *(const v8bf*)(p + 8);
    return pack16(lo, hi);
}

// B fragment (32x16 bf16): lane-half holds 16 contiguous K values of column n
__device__ __forceinline__ v16bf load_frag_contig16(const bf16* p) {
    v8bf lo = *(const v8bf*)p;
    v8bf hi = *(const v8bf*)(p + 8);
    return pack16(lo, hi);
}

#if MLA_HAS_ASYNC
// one 16-byte global->LDS async copy (tracked by ASYNCcnt)
__device__ __forceinline__ void async_copy16(const bf16* g, bf16* l) {
    __builtin_amdgcn_global_load_async_to_lds_b128(
        (AS1 v4i_t*)g, (AS3 v4i_t*)l, 0, 0);
}
__device__ __forceinline__ void async_wait0() {
    __builtin_amdgcn_s_wait_asynccnt(0);
}
#else
__device__ __forceinline__ void async_copy16(const bf16* g, bf16* l) {
    *(v8bf*)l = *(const v8bf*)g;
}
__device__ __forceinline__ void async_wait0() {}
#endif

// ---------------------------------------------------------------------------
// fp32 -> bf16 conversion (row-major preserved; used for x)
// ---------------------------------------------------------------------------
__global__ void mla_cvt_f32_bf16(const float* __restrict__ src,
                                 bf16* __restrict__ dst, int n) {
    int i = blockIdx.x * blockDim.x + threadIdx.x;
    if (i < n) dst[i] = (bf16)src[i];
}

// ---------------------------------------------------------------------------
// fp32 [K][N] -> bf16 [N][K] transposed conversion (LDS 32x32 tiles, both
// global sides coalesced). One-time pass per weight matrix.
// ---------------------------------------------------------------------------
__global__ void __launch_bounds__(256)
mla_cvt_transpose(const float* __restrict__ src, bf16* __restrict__ dst,
                  int K, int N) {
    __shared__ float tile[32][33];
    const int k0 = blockIdx.y * 32;
    const int n0 = blockIdx.x * 32;
    const int tx = threadIdx.x & 31;
    const int ty = threadIdx.x >> 5;      // 0..7
#pragma unroll
    for (int i = 0; i < 32; i += 8)
        tile[ty + i][tx] = src[(size_t)(k0 + ty + i) * N + n0 + tx];
    __syncthreads();
#pragma unroll
    for (int i = 0; i < 32; i += 8)
        dst[(size_t)(n0 + ty + i) * K + k0 + tx] = (bf16)tile[tx][ty + i];
}

// ---------------------------------------------------------------------------
// Generic GEMM: C[M,N] = A[M,K] * Bt[N,K]^T, A/Bt bf16 row-major.
// 128x128 block tile, 256 threads = 8 waves in a 4(M) x 2(N) grid; each wave
// computes a 32x64 sub-tile: 2 A-frags x 4 B-frags = 8 WMMAs per K-step.
// Both operands staged with contiguous async-to-LDS copies; double-buffered.
// ---------------------------------------------------------------------------
__device__ __forceinline__ void gemm_stage(
    const bf16* __restrict__ A, const bf16* __restrict__ Bt,
    bf16 (*Abuf)[32], bf16 (*Bbuf)[32],
    int tid, int m0, int n0, int K, int kb) {
    // A tile 128x32 = 512 x 16B, 2 per thread
#pragma unroll
    for (int v = 0; v < 2; ++v) {
        int idx = tid + v * 256;
        int r   = idx >> 2;
        int c   = (idx & 3) * 8;
        async_copy16(&A[(size_t)(m0 + r) * K + kb + c], &Abuf[r][c]);
    }
    // B tile (transposed layout) 128x32 = 512 x 16B, 2 per thread
#pragma unroll
    for (int v = 0; v < 2; ++v) {
        int idx = tid + v * 256;
        int r   = idx >> 2;
        int c   = (idx & 3) * 8;
        async_copy16(&Bt[(size_t)(n0 + r) * K + kb + c], &Bbuf[r][c]);
    }
}

__global__ void __launch_bounds__(256)
mla_gemm_bf16_wmma(const bf16* __restrict__ A, const bf16* __restrict__ Bt,
                   bf16* __restrict__ Cb, float* __restrict__ Cf,
                   int M, int N, int K) {
    __shared__ __attribute__((aligned(16))) bf16 Alds[2][128][32];  // 16 KB
    __shared__ __attribute__((aligned(16))) bf16 Btl[2][128][32];   // 16 KB

    const int tid  = threadIdx.x;
    const int lane = tid & 31;
    const int wid  = tid >> 5;
    const int wy   = wid >> 1;      // 0..3  (M direction, 32 rows each)
    const int wx   = wid & 1;       // 0..1  (N direction, 64 cols each)
    const int m0   = blockIdx.y * 128;
    const int n0   = blockIdx.x * 128;

    const int am    = lane & 15;
    const int ak0   = (lane >> 4) * 16;
    const int bn    = lane & 15;
    const int bk0   = (lane >> 4) * 16;
    const int mhalf = (lane >> 4) * 8;

    v8f acc[2][4];
#pragma unroll
    for (int mi = 0; mi < 2; ++mi)
#pragma unroll
        for (int ni = 0; ni < 4; ++ni) acc[mi][ni] = (v8f){};

    gemm_stage(A, Bt, Alds[0], Btl[0], tid, m0, n0, K, 0);
    async_wait0();
    __syncthreads();

    int buf = 0;
    for (int kb = 0; kb < K; kb += 32) {
        if (kb + 32 < K)
            gemm_stage(A, Bt, Alds[buf ^ 1], Btl[buf ^ 1], tid, m0, n0, K, kb + 32);
        if (kb + 64 < K)   // L2 prefetch two slabs ahead (global_prefetch_b8)
            __builtin_prefetch(&A[(size_t)(m0 + (tid >> 1)) * K + kb + 64 + (tid & 1) * 16], 0, 1);

        v16bf af[2], bfv[4];
#pragma unroll
        for (int mi = 0; mi < 2; ++mi)
            af[mi] = load_frag_interleave8(&Alds[buf][wy * 32 + mi * 16 + am][ak0]);
#pragma unroll
        for (int ni = 0; ni < 4; ++ni)
            bfv[ni] = load_frag_contig16(&Btl[buf][wx * 64 + ni * 16 + bn][bk0]);
#pragma unroll
        for (int mi = 0; mi < 2; ++mi)
#pragma unroll
            for (int ni = 0; ni < 4; ++ni)
                acc[mi][ni] = __builtin_amdgcn_wmma_f32_16x16x32_bf16(
                    false, af[mi], false, bfv[ni], (short)0, acc[mi][ni], false, false);

        async_wait0();
        __syncthreads();
        buf ^= 1;
    }

    // C layout: VGPR r -> row = r + 8*(lane>=16), col = lane&15
    const int row0 = m0 + wy * 32 + mhalf;
    const int col0 = n0 + wx * 64 + (lane & 15);
    if (Cf) {
#pragma unroll
        for (int mi = 0; mi < 2; ++mi)
#pragma unroll
            for (int ni = 0; ni < 4; ++ni)
#pragma unroll
                for (int r = 0; r < 8; ++r)
                    Cf[(size_t)(row0 + mi * 16 + r) * N + col0 + ni * 16] = acc[mi][ni][r];
    } else {
#pragma unroll
        for (int mi = 0; mi < 2; ++mi)
#pragma unroll
            for (int ni = 0; ni < 4; ++ni)
#pragma unroll
                for (int r = 0; r < 8; ++r)
                    Cb[(size_t)(row0 + mi * 16 + r) * N + col0 + ni * 16] =
                        (bf16)acc[mi][ni][r];
    }
}

// ---------------------------------------------------------------------------
// Assemble q or k: content [B*T][H*128] + rope-proj [B*T][H*64]
//   -> out [B][H][T][192] with NeoX RoPE applied to the last 64 dims.
// ---------------------------------------------------------------------------
__global__ void mla_assemble_rope(const bf16* __restrict__ content,
                                  const bf16* __restrict__ rope,
                                  bf16* __restrict__ out) {
    size_t idx = (size_t)blockIdx.x * blockDim.x + threadIdx.x;
    const size_t total = (size_t)B_SZ * N_HEADS * T_SZ * D_QK;
    if (idx >= total) return;
    int d = (int)(idx % D_QK);
    size_t tmp = idx / D_QK;
    int t = (int)(tmp % T_SZ); tmp /= T_SZ;
    int h = (int)(tmp % N_HEADS);
    int b = (int)(tmp / N_HEADS);
    size_t row = (size_t)b * T_SZ + t;

    float val;
    if (d < D_HEAD) {
        val = (float)content[row * (size_t)(N_HEADS * D_HEAD) + h * D_HEAD + d];
    } else {
        int i = d - D_HEAD;
        const bf16* rp = rope + row * (size_t)(N_HEADS * ROPE_DIM) + h * ROPE_DIM;
        int j = (i < 32) ? i : i - 32;
        float inv = __powf(10000.0f, -2.0f * (float)j / (float)ROPE_DIM);
        float ang = (float)t * inv;
        float c = __cosf(ang), s = __sinf(ang);
        float xv = (float)rp[i];
        float xr = (i < 32) ? -(float)rp[i + 32] : (float)rp[i - 32];
        val = xv * c + xr * s;
    }
    out[idx] = (bf16)val;
}

// v: [B*T][H*128] -> d-major [B][H][128][T] (so attention V^T staging is
// contiguous async copies)
__global__ void mla_gather_v(const bf16* __restrict__ vin,
                             bf16* __restrict__ vout) {
    size_t idx = (size_t)blockIdx.x * blockDim.x + threadIdx.x;
    const size_t total = (size_t)B_SZ * N_HEADS * T_SZ * D_HEAD;
    if (idx >= total) return;
    // output index decomposition: [b][h][d][t]
    int t = (int)(idx % T_SZ);
    size_t tmp = idx / T_SZ;
    int d = (int)(tmp % D_HEAD); tmp /= D_HEAD;
    int h = (int)(tmp % N_HEADS);
    int b = (int)(tmp / N_HEADS);
    vout[idx] = vin[((size_t)b * T_SZ + t) * (size_t)(N_HEADS * D_HEAD)
                    + h * D_HEAD + d];
}

// ---------------------------------------------------------------------------
// Flash attention: per (b, h, 64-row query tile). 128 threads = 4 waves;
// each wave owns 16 query rows. Online softmax, causal mask.
//   q,k: [B][H][T][192], v: [B][H][128][T] (d-major), out: [B*T][H*128] bf16
// ---------------------------------------------------------------------------
__global__ void __launch_bounds__(128)
mla_flash_attn(const bf16* __restrict__ Q, const bf16* __restrict__ Kh,
               const bf16* __restrict__ V, bf16* __restrict__ Oout) {
    __shared__ __attribute__((aligned(16))) bf16 Klds[64][D_QK];   // 24 KB
    __shared__ __attribute__((aligned(16))) bf16 Vt[D_HEAD][64];   // 16 KB (V^T)
    __shared__ __attribute__((aligned(16))) bf16 Plds[4][16][64];  //  8 KB

    const int tid  = threadIdx.x;
    const int lane = tid & 31;
    const int w    = tid >> 5;          // wave 0..3
    const int qt   = blockIdx.x;        // query tile (64 rows)
    const int h    = blockIdx.y;
    const int b    = blockIdx.z;

    const size_t hoff_qk = ((size_t)(b * N_HEADS + h)) * T_SZ * D_QK;
    const size_t hoff_v  = ((size_t)(b * N_HEADS + h)) * (size_t)D_HEAD * T_SZ;
    const bf16* q  = Q  + hoff_qk;
    const bf16* k  = Kh + hoff_qk;
    const bf16* vt = V  + hoff_v;       // [128][T] rows

    const int qrow0 = qt * 64 + w * 16;
    const int am    = lane & 15;
    const int ak0   = (lane >> 4) * 16;
    const int bn    = lane & 15;
    const int bk0   = (lane >> 4) * 16;
    const int mhalf = (lane >> 4) * 8;

    // resident Q fragments: 6 chunks of K=32 over d=192
    v16bf qf[6];
#pragma unroll
    for (int kd = 0; kd < 6; ++kd) {
        const bf16* qrow = q + (size_t)(qrow0 + am) * D_QK + kd * 32;
        qf[kd] = load_frag_interleave8(qrow + ak0);
    }

    v8f of[8];
#pragma unroll
    for (int dt = 0; dt < 8; ++dt) of[dt] = (v8f){};
    float mrow[8], lrow[8];
#pragma unroll
    for (int r = 0; r < 8; ++r) { mrow[r] = -3.0e38f; lrow[r] = 0.0f; }

    const float scale = 0.07216878364870322f;   // 1/sqrt(192)

    for (int kt = 0; kt <= qt; ++kt) {
        // ---- stage K tile [64][192]: 12 x 16B async copies per thread ----
        {
            int base = tid * 12;
#pragma unroll
            for (int i = 0; i < 12; ++i) {
                int vi = base + i;            // 0..1535
                int r  = vi / 24;
                int c  = (vi % 24) * 8;
                async_copy16(&k[((size_t)(kt * 64 + r)) * D_QK + c], &Klds[r][c]);
            }
        }
        // ---- stage V^T tile [128][64]: 8 x 16B async copies per thread ----
        {
            int base = tid * 8;
#pragma unroll
            for (int i = 0; i < 8; ++i) {
                int vi = base + i;            // 0..1023
                int r  = vi >> 3;             // value dim 0..127
                int c  = (vi & 7) * 8;        // 0..56
                async_copy16(&vt[(size_t)r * T_SZ + kt * 64 + c], &Vt[r][c]);
            }
        }
        async_wait0();
        __syncthreads();

        // ---- S = Q * K^T : 16 rows x 64 keys per wave ----
        v8f s[4];
#pragma unroll
        for (int nt = 0; nt < 4; ++nt) {
            v8f sacc = {};
#pragma unroll
            for (int kd = 0; kd < 6; ++kd) {
                v16bf bfr = load_frag_contig16(&Klds[nt * 16 + bn][kd * 32 + bk0]);
                sacc = __builtin_amdgcn_wmma_f32_16x16x32_bf16(
                           false, qf[kd], false, bfr, (short)0, sacc, false, false);
            }
            s[nt] = sacc;
        }

        // scale + causal mask (diagonal tile only)
        if (kt == qt) {
#pragma unroll
            for (int nt = 0; nt < 4; ++nt) {
#pragma unroll
                for (int r = 0; r < 8; ++r) {
                    int kglob = kt * 64 + nt * 16 + (lane & 15);
                    int qglob = qrow0 + mhalf + r;
                    s[nt][r] = (kglob > qglob) ? -3.0e38f : s[nt][r] * scale;
                }
            }
        } else {
#pragma unroll
            for (int nt = 0; nt < 4; ++nt)
#pragma unroll
                for (int r = 0; r < 8; ++r) s[nt][r] *= scale;
        }

        // ---- online softmax (row reductions across the 16-lane half) ----
        float fac[8];
#pragma unroll
        for (int r = 0; r < 8; ++r) {
            float mx = -3.0e38f;
#pragma unroll
            for (int nt = 0; nt < 4; ++nt) mx = fmaxf(mx, s[nt][r]);
#pragma unroll
            for (int off = 8; off > 0; off >>= 1)
                mx = fmaxf(mx, __shfl_xor(mx, off, 32));
            float mnew = fmaxf(mrow[r], mx);
            fac[r] = __expf(mrow[r] - mnew);
            float sum = 0.0f;
#pragma unroll
            for (int nt = 0; nt < 4; ++nt) {
                float p = __expf(s[nt][r] - mnew);
                s[nt][r] = p;
                sum += p;
            }
#pragma unroll
            for (int off = 8; off > 0; off >>= 1)
                sum += __shfl_xor(sum, off, 32);
            lrow[r] = lrow[r] * fac[r] + sum;
            mrow[r] = mnew;
        }
        // rescale running output
#pragma unroll
        for (int dt = 0; dt < 8; ++dt)
#pragma unroll
            for (int r = 0; r < 8; ++r) of[dt][r] *= fac[r];

        // ---- P (C-layout) -> LDS (A-layout source), wave-private ----
#pragma unroll
        for (int nt = 0; nt < 4; ++nt)
#pragma unroll
            for (int r = 0; r < 8; ++r)
                Plds[w][mhalf + r][nt * 16 + (lane & 15)] = (bf16)s[nt][r];

        // ---- O += P * V ----
#pragma unroll
        for (int dt = 0; dt < 8; ++dt) {
            v8f oacc = of[dt];
#pragma unroll
            for (int kk = 0; kk < 2; ++kk) {
                v16bf pa = load_frag_interleave8(&Plds[w][am][kk * 32 + ak0]);
                v16bf vb = load_frag_contig16(&Vt[dt * 16 + bn][kk * 32 + bk0]);
                oacc = __builtin_amdgcn_wmma_f32_16x16x32_bf16(
                           false, pa, false, vb, (short)0, oacc, false, false);
            }
            of[dt] = oacc;
        }
        __syncthreads();   // protect Klds/Vt before next stage
    }

    // ---- epilogue: normalize and scatter to [B*T][H*128] ----
#pragma unroll
    for (int dt = 0; dt < 8; ++dt) {
#pragma unroll
        for (int r = 0; r < 8; ++r) {
            int t = qrow0 + mhalf + r;
            float val = of[dt][r] / lrow[r];
            Oout[((size_t)b * T_SZ + t) * D_MODEL + h * D_HEAD + dt * 16 + (lane & 15)]
                = (bf16)val;
        }
    }
}

// ---------------------------------------------------------------------------
// host launch
// ---------------------------------------------------------------------------
extern "C" void kernel_launch(void* const* d_in, const int* in_sizes, int n_in,
                              void* d_out, int out_size, void* d_ws, size_t ws_size,
                              hipStream_t stream) {
    (void)in_sizes; (void)n_in; (void)out_size; (void)ws_size;

    const float* x    = (const float*)d_in[0];
    const float* Wdq  = (const float*)d_in[1];
    const float* Wuq  = (const float*)d_in[2];
    const float* Wqr  = (const float*)d_in[3];
    const float* Wdkv = (const float*)d_in[4];
    const float* Wuk  = (const float*)d_in[5];
    const float* Wuv  = (const float*)d_in[6];
    const float* Wkr  = (const float*)d_in[7];
    const float* Wo   = (const float*)d_in[8];
    float* out = (float*)d_out;

    char* ws = (char*)d_ws;
    size_t off = 0;
    auto alloc_bf = [&](size_t elems) -> bf16* {
        off = (off + 255) & ~(size_t)255;
        bf16* p = (bf16*)(ws + off);
        off += elems * sizeof(bf16);
        return p;
    };

    bf16* xb   = alloc_bf((size_t)M_ROWS * D_MODEL);
    bf16* wdq  = alloc_bf((size_t)D_MODEL * Q_COMP);     // transposed [N][K]
    bf16* wuq  = alloc_bf((size_t)Q_COMP * D_MODEL);
    bf16* wqr  = alloc_bf((size_t)Q_COMP * (N_HEADS * ROPE_DIM));
    bf16* wdkv = alloc_bf((size_t)D_MODEL * KV_COMP);
    bf16* wuk  = alloc_bf((size_t)KV_COMP * D_MODEL);
    bf16* wuv  = alloc_bf((size_t)KV_COMP * D_MODEL);
    bf16* wkr  = alloc_bf((size_t)KV_COMP * (N_HEADS * ROPE_DIM));
    bf16* wo   = alloc_bf((size_t)D_MODEL * D_MODEL);
    bf16* cq   = alloc_bf((size_t)M_ROWS * Q_COMP);
    bf16* ckv  = alloc_bf((size_t)M_ROWS * KV_COMP);
    bf16* tmpA = alloc_bf((size_t)M_ROWS * D_MODEL);               // content / v_tmp / attn-out
    bf16* tmpB = alloc_bf((size_t)M_ROWS * (N_HEADS * ROPE_DIM));  // rope proj
    bf16* qbuf = alloc_bf((size_t)B_SZ * N_HEADS * T_SZ * D_QK);
    bf16* kbuf = alloc_bf((size_t)B_SZ * N_HEADS * T_SZ * D_QK);
    bf16* vbuf = alloc_bf((size_t)B_SZ * N_HEADS * T_SZ * D_HEAD); // d-major

    auto cvt = [&](const float* src, bf16* dst, size_t n) {
        mla_cvt_f32_bf16<<<dim3((unsigned)((n + 255) / 256)), dim3(256), 0, stream>>>(
            src, dst, (int)n);
    };
    // fp32 W[K][N] -> bf16 Wt[N][K]
    auto cvtT = [&](const float* src, bf16* dst, int K, int N) {
        mla_cvt_transpose<<<dim3(N / 32, K / 32), dim3(256), 0, stream>>>(
            src, dst, K, N);
    };
    auto gemm = [&](const bf16* A, const bf16* Bt, bf16* Cb, float* Cf,
                    int M, int N, int K) {
        mla_gemm_bf16_wmma<<<dim3(N / 128, M / 128), dim3(256), 0, stream>>>(
            A, Bt, Cb, Cf, M, N, K);
    };

    // 1) convert: x row-major, weights transposed
    cvt (x,    xb, (size_t)M_ROWS * D_MODEL);
    cvtT(Wdq,  wdq,  D_MODEL, Q_COMP);
    cvtT(Wuq,  wuq,  Q_COMP,  D_MODEL);
    cvtT(Wqr,  wqr,  Q_COMP,  N_HEADS * ROPE_DIM);
    cvtT(Wdkv, wdkv, D_MODEL, KV_COMP);
    cvtT(Wuk,  wuk,  KV_COMP, D_MODEL);
    cvtT(Wuv,  wuv,  KV_COMP, D_MODEL);
    cvtT(Wkr,  wkr,  KV_COMP, N_HEADS * ROPE_DIM);
    cvtT(Wo,   wo,   D_MODEL, D_MODEL);

    // 2) compression GEMMs
    gemm(xb, wdq,  cq,  nullptr, M_ROWS, Q_COMP,  D_MODEL);
    gemm(xb, wdkv, ckv, nullptr, M_ROWS, KV_COMP, D_MODEL);

    // 3) query path: up-projections + RoPE assembly
    gemm(cq, wuq, tmpA, nullptr, M_ROWS, D_MODEL, Q_COMP);
    gemm(cq, wqr, tmpB, nullptr, M_ROWS, N_HEADS * ROPE_DIM, Q_COMP);
    {
        size_t n = (size_t)B_SZ * N_HEADS * T_SZ * D_QK;
        mla_assemble_rope<<<dim3((unsigned)((n + 255) / 256)), dim3(256), 0, stream>>>(
            tmpA, tmpB, qbuf);
    }

    // 4) key path
    gemm(ckv, wuk, tmpA, nullptr, M_ROWS, D_MODEL, KV_COMP);
    gemm(ckv, wkr, tmpB, nullptr, M_ROWS, N_HEADS * ROPE_DIM, KV_COMP);
    {
        size_t n = (size_t)B_SZ * N_HEADS * T_SZ * D_QK;
        mla_assemble_rope<<<dim3((unsigned)((n + 255) / 256)), dim3(256), 0, stream>>>(
            tmpA, tmpB, kbuf);
    }

    // 5) value path (then repack d-major)
    gemm(ckv, wuv, tmpA, nullptr, M_ROWS, D_MODEL, KV_COMP);
    {
        size_t n = (size_t)B_SZ * N_HEADS * T_SZ * D_HEAD;
        mla_gather_v<<<dim3((unsigned)((n + 255) / 256)), dim3(256), 0, stream>>>(
            tmpA, vbuf);
    }

    // 6) fused causal flash attention -> tmpA as [B*T][H*128]
    mla_flash_attn<<<dim3(T_SZ / 64, N_HEADS, B_SZ), dim3(128), 0, stream>>>(
        qbuf, kbuf, vbuf, tmpA);

    // 7) output projection (fp32 result)
    gemm(tmpA, wo, nullptr, out, M_ROWS, D_MODEL, D_MODEL);
}
